// Multi_Head_Attention_62130996904465
// MI455X (gfx1250) — compile-verified, matrix operand order checked by reading, not run on hardware
//
#include <hip/hip_runtime.h>

// MHA forward for MI455X (gfx1250): bf16 WMMA everywhere, f32 accumulate.
// B=4, S=2048, H=1024, heads=16, d=64.

typedef __attribute__((ext_vector_type(16))) __bf16 v16bf;
typedef __attribute__((ext_vector_type(8)))  __bf16 v8bf;
typedef __attribute__((ext_vector_type(8)))  float  v8f;

#define WMMA_BF16(a, b, c) \
  __builtin_amdgcn_wmma_f32_16x16x32_bf16(false, (a), false, (b), (short)0, (c), false, false)

#define BB   4
#define SS   2048
#define HH   1024
#define NH   16
#define DD   64

// Load a 16x32 bf16 fragment (A layout, ISA 7.12.2) from row-major memory.
// Lane L: row M = L&15; K-chunks {0..7,16..23} (L<16) or {8..15,24..31} (L>=16).
// The same loader serves B operands when B's columns are contiguous rows in memory.
static __device__ inline v16bf load_frag(const __bf16* base, int rowStride, int lane, int kOff) {
  int m  = lane & 15;
  int kb = kOff + ((lane >> 4) << 3);
  const __bf16* p = base + (size_t)m * rowStride + kb;
  v8bf lo = *(const v8bf*)(p);
  v8bf hi = *(const v8bf*)(p + 16);
  return __builtin_shufflevector(lo, hi, 0,1,2,3,4,5,6,7,8,9,10,11,12,13,14,15);
}

__global__ void cvt_f32_bf16(const float* __restrict__ in, __bf16* __restrict__ out, int n) {
  int i = blockIdx.x * blockDim.x + threadIdx.x;
  int stride = gridDim.x * blockDim.x;
  for (; i < n; i += stride) out[i] = (__bf16)in[i];
}

// Q/K/V projection: out = x @ W.T + b.  blockIdx.y selects q/k/v.
// Each wave computes a 32x32 output group (2x2 WMMA tiles) so every A/B
// fragment is reused twice: 4 fragment loads feed 4 WMMAs per k-step.
// Q,K stored bf16 [B,h,S,d]; V stored transposed bf16 [B,h,d,S].
__global__ void qkv_proj_kernel(const __bf16* __restrict__ xb,
                                const __bf16* __restrict__ Wq, const float* __restrict__ bq, __bf16* __restrict__ Qo,
                                const __bf16* __restrict__ Wk, const float* __restrict__ bk, __bf16* __restrict__ Ko,
                                const __bf16* __restrict__ Wv, const float* __restrict__ bv, __bf16* __restrict__ Vt) {
  const int lane  = threadIdx.x & 31;
  const int wave  = threadIdx.x >> 5;
  const int which = blockIdx.y;
  const __bf16* W    = (which == 0) ? Wq : (which == 1) ? Wk : Wv;
  const float*  bias = (which == 0) ? bq : (which == 1) ? bk : bv;

  // 8192 groups: 256 row-groups (32 rows) x 32 col-groups (32 cols)
  const int g  = blockIdx.x * 8 + wave;
  const int mg = g >> 5;
  const int ng = g & 31;

  const __bf16* A0 = xb + (size_t)mg * 32 * HH;  // rows of x, stride H
  const __bf16* A1 = A0 + 16 * HH;
  const __bf16* B0 = W  + (size_t)ng * 32 * HH;  // rows of W = cols of W.T, stride H
  const __bf16* B1 = B0 + 16 * HH;

  v8f acc[2][2] = {{{}, {}}, {{}, {}}};
  #pragma unroll 2
  for (int kc = 0; kc < HH / 32; ++kc) {
    const int ko = kc * 32;
    v16bf a0 = load_frag(A0, HH, lane, ko);
    v16bf a1 = load_frag(A1, HH, lane, ko);
    v16bf b0 = load_frag(B0, HH, lane, ko);
    v16bf b1 = load_frag(B1, HH, lane, ko);
    acc[0][0] = WMMA_BF16(a0, b0, acc[0][0]);
    acc[0][1] = WMMA_BF16(a0, b1, acc[0][1]);
    acc[1][0] = WMMA_BF16(a1, b0, acc[1][0]);
    acc[1][1] = WMMA_BF16(a1, b1, acc[1][1]);
  }

  #pragma unroll
  for (int mi = 0; mi < 2; ++mi) {
    #pragma unroll
    for (int ni = 0; ni < 2; ++ni) {
      const int n    = ng * 32 + ni * 16 + (lane & 15);  // output column (lane-fixed)
      const float bn = bias[n];
      const int head = n >> 6;
      const int dd   = n & 63;
      const int r0   = mg * 32 + mi * 16 + ((lane >> 4) << 3);  // first row this lane holds
      const int b_   = r0 >> 11;                                // r0 / S
      const int s0   = r0 & (SS - 1);

      if (which <= 1) {
        __bf16* dst = (which == 0) ? Qo : Ko;
        #pragma unroll
        for (int i = 0; i < 8; ++i) {
          int s = s0 + i;
          dst[(((size_t)b_ * NH + head) * SS + s) * DD + dd] = (__bf16)(acc[mi][ni][i] + bn);
        }
      } else {
        v8bf pkt;
        #pragma unroll
        for (int i = 0; i < 8; ++i) pkt[i] = (__bf16)(acc[mi][ni][i] + bn);
        *(v8bf*)(Vt + (((size_t)b_ * NH + head) * DD + dd) * SS + s0) = pkt;  // contiguous in s
      }
    }
  }
}

// One workgroup = 8 waves = one (b, head, 16-row query tile).
// Full score row (16 x 2048 f32 = 128KB) lives in LDS (MI455X: 320KB/WGP);
// softmax in-place; P read back as bf16 A-fragments for the PV WMMAs.
__global__ void attention_kernel(const __bf16* __restrict__ Qb,
                                 const __bf16* __restrict__ Kb,
                                 const __bf16* __restrict__ Vt,
                                 const float*  __restrict__ atten,
                                 float* __restrict__ out) {
  extern __shared__ float smem[];
  float* sc      = smem;              // [16][2048] scores -> exp(scores)
  float* attnRow = sc + 16 * SS;      // [16] padding-mask values per query row
  float* rinv    = attnRow + 16;      // [16] 1/rowsum

  const int lane = threadIdx.x & 31;
  const int wave = threadIdx.x >> 5;
  const int qt   = blockIdx.x;        // 0..127
  const int head = blockIdx.y;        // 0..15
  const int b    = blockIdx.z;        // 0..3
  const int bh   = b * NH + head;

  if (threadIdx.x < 16)
    attnRow[threadIdx.x] = atten[b * SS + qt * 16 + threadIdx.x];

  const __bf16* Qbase = Qb + ((size_t)bh * SS + qt * 16) * DD;
  v16bf qa0 = load_frag(Qbase, DD, lane, 0);
  v16bf qa1 = load_frag(Qbase, DD, lane, 32);

  const float scale = 0.125f;  // 1/sqrt(64)

  // ---- phase 1: score tiles -> LDS (causal masked) ----
  for (int kt = wave; kt < SS / 16; kt += 8) {
    const int col = kt * 16 + (lane & 15);
    if (kt > qt) {  // tile entirely above the diagonal: all masked
      #pragma unroll
      for (int i = 0; i < 8; ++i) {
        int m = i + ((lane >> 4) << 3);
        sc[m * SS + col] = -1e9f;
      }
      continue;
    }
    const __bf16* Kt = Kb + ((size_t)bh * SS + kt * 16) * DD;
    v16bf k0 = load_frag(Kt, DD, lane, 0);
    v16bf k1 = load_frag(Kt, DD, lane, 32);
    v8f c = {};
    c = WMMA_BF16(qa0, k0, c);
    c = WMMA_BF16(qa1, k1, c);
    #pragma unroll
    for (int i = 0; i < 8; ++i) {
      int m = i + ((lane >> 4) << 3);
      int q = qt * 16 + m;
      sc[m * SS + col] = (col > q) ? -1e9f : c[i] * scale;
    }
  }
  __syncthreads();

  // ---- phase 2: row softmax (wave w owns rows 2w, 2w+1), exp in place ----
  for (int rr = 0; rr < 2; ++rr) {
    const int m = wave * 2 + rr;
    float* row = sc + m * SS;
    const bool rowMasked = attnRow[m] >= 1.0f;  // padded query row -> uniform softmax
    float sum;
    if (!rowMasked) {
      float mx = -3.4e38f;
      for (int c0 = lane; c0 < SS; c0 += 32) mx = fmaxf(mx, row[c0]);
      #pragma unroll
      for (int off = 16; off; off >>= 1) mx = fmaxf(mx, __shfl_xor(mx, off, 32));
      float s = 0.f;
      for (int c0 = lane; c0 < SS; c0 += 32) {
        float e = __expf(row[c0] - mx);
        row[c0] = e;
        s += e;
      }
      #pragma unroll
      for (int off = 16; off; off >>= 1) s += __shfl_xor(s, off, 32);
      sum = s;
    } else {
      for (int c0 = lane; c0 < SS; c0 += 32) row[c0] = 1.0f;
      sum = (float)SS;
    }
    if (lane == 0) rinv[m] = 1.0f / sum;
  }
  __syncthreads();

  // ---- phase 3: out = (P/rowsum) @ V ; 4 waves, one 16-col slice of d each ----
  if (wave < 4) {
    const __bf16* Vbase = Vt + ((size_t)bh * DD + wave * 16) * SS;  // rows of V^T, stride S
    v8f acc = {};
    for (int kc = 0; kc < SS / 32; ++kc) {
      // A fragment of P: read f32 from LDS, convert to bf16 in-register
      const int m  = lane & 15;
      const int kb = kc * 32 + ((lane >> 4) << 3);
      const float* p = sc + m * SS + kb;
      float4 f0 = *(const float4*)(p);
      float4 f1 = *(const float4*)(p + 4);
      float4 f2 = *(const float4*)(p + 16);
      float4 f3 = *(const float4*)(p + 20);
      v16bf a;
      a[0]=(__bf16)f0.x;  a[1]=(__bf16)f0.y;  a[2]=(__bf16)f0.z;  a[3]=(__bf16)f0.w;
      a[4]=(__bf16)f1.x;  a[5]=(__bf16)f1.y;  a[6]=(__bf16)f1.z;  a[7]=(__bf16)f1.w;
      a[8]=(__bf16)f2.x;  a[9]=(__bf16)f2.y;  a[10]=(__bf16)f2.z; a[11]=(__bf16)f2.w;
      a[12]=(__bf16)f3.x; a[13]=(__bf16)f3.y; a[14]=(__bf16)f3.z; a[15]=(__bf16)f3.w;
      v16bf bb = load_frag(Vbase, SS, lane, kc * 32);
      acc = WMMA_BF16(a, bb, acc);
    }
    const int n = wave * 16 + (lane & 15);
    #pragma unroll
    for (int i = 0; i < 8; ++i) {
      int m = i + ((lane >> 4) << 3);
      int q = qt * 16 + m;
      out[((size_t)b * SS + q) * HH + head * DD + n] = acc[i] * rinv[m];
    }
  }
}

extern "C" void kernel_launch(void* const* d_in, const int* in_sizes, int n_in,
                              void* d_out, int out_size, void* d_ws, size_t ws_size,
                              hipStream_t stream) {
  const float* x     = (const float*)d_in[0];
  const float* Wq    = (const float*)d_in[1];
  const float* bq    = (const float*)d_in[2];
  const float* Wk    = (const float*)d_in[3];
  const float* bk    = (const float*)d_in[4];
  const float* Wv    = (const float*)d_in[5];
  const float* bv    = (const float*)d_in[6];
  const float* atten = (const float*)d_in[7];
  float* out = (float*)d_out;

  // workspace layout (bytes), total ~73.4 MB
  char* ws = (char*)d_ws;
  __bf16* xb  = (__bf16*)(ws);                       // 4*2048*1024*2 = 16 MB
  __bf16* Wqb = (__bf16*)(ws + 16777216);            // 2 MB
  __bf16* Wkb = (__bf16*)(ws + 18874368);            // 2 MB
  __bf16* Wvb = (__bf16*)(ws + 20971520);            // 2 MB
  __bf16* Qb  = (__bf16*)(ws + 23068672);            // 16 MB  [B,h,S,d]
  __bf16* Kb  = (__bf16*)(ws + 39845888);            // 16 MB  [B,h,S,d]
  __bf16* Vt  = (__bf16*)(ws + 56623104);            // 16 MB  [B,h,d,S]

  cvt_f32_bf16<<<4096, 256, 0, stream>>>(x,  xb,  BB * SS * HH);
  cvt_f32_bf16<<<1024, 256, 0, stream>>>(Wq, Wqb, HH * HH);
  cvt_f32_bf16<<<1024, 256, 0, stream>>>(Wk, Wkb, HH * HH);
  cvt_f32_bf16<<<1024, 256, 0, stream>>>(Wv, Wvb, HH * HH);

  // 256 row-groups * 32 col-groups = 8192 groups, 8 waves/block -> 1024 blocks; y = q/k/v
  qkv_proj_kernel<<<dim3(1024, 3), 256, 0, stream>>>(xb, Wqb, bq, Qb, Wkb, bk, Kb, Wvb, bv, Vt);

  const size_t smem = (size_t)(16 * SS + 32) * sizeof(float);  // 128 KB + tails
  attention_kernel<<<dim3(SS / 16, NH, BB), 256, smem, stream>>>(Qb, Kb, Vt, atten, out);
}
